// GATClassifier_37718402793667
// MI455X (gfx1250) — matrix operand order repaired and checked by reading, not compiled, not run on hardware
//
#include <hip/hip_runtime.h>
#include <hip/hip_bf16.h>
#include <stddef.h>

// ---------------- problem constants (structure of the model) ----------------
#define DIMK   128      // input feature dim (K of the GEMM)
#define NH     5        // heads
#define NC     10       // channels per head
#define HC     50       // NH*NC
#define NGRAPH 256      // graphs in batch
#define NEG_SLOPE 0.2f

typedef __attribute__((ext_vector_type(16))) _Float16 v16h;
typedef __attribute__((ext_vector_type(8)))  float    v8f;
typedef int v4i __attribute__((vector_size(16)));   // matches async-LDS builtin param type

// ---------------- helpers ----------------
__device__ __forceinline__ float leaky_relu(float v) {
    return v > 0.0f ? v : NEG_SLOPE * v;
}

// float atomic max via monotonic integer mapping (native int atomics, no CAS loop)
__device__ __forceinline__ void atomicMaxFloat(float* addr, float val) {
    if (val >= 0.0f) {
        atomicMax((int*)addr, __float_as_int(val));
    } else {
        atomicMin((unsigned int*)addr, (unsigned int)__float_as_uint(val));
    }
}

// ---------------- kernel 0: init scratch ----------------
__global__ __launch_bounds__(256) void k_init(float* mmax, float* denom, float* acc,
                                              float* gsum, float* cnt, int N) {
    int i = blockIdx.x * blockDim.x + threadIdx.x;
    int NH5 = N * NH;
    int NHC = N * HC;
    if (i < NH5)        { mmax[i] = -__builtin_inff(); denom[i] = 0.0f; }
    if (i < NHC)        acc[i]  = 0.0f;
    if (i < NGRAPH*HC)  gsum[i] = 0.0f;
    if (i < NGRAPH)     cnt[i]  = 0.0f;
}

// ---------------- kernel 1: xw = x @ W via v_wmma_f32_16x16x32_f16 ----------------
// One wave computes a 16-row x 64-col tile; block = 8 waves = 128 rows.
// Stage 1: W (128x50 f32) is async-DMAed into LDS (GLOBAL_LOAD_ASYNC_TO_LDS_B128).
// Stage 2: converted f32->f16 into the exact per-lane B-fragment layout so each
//          fragment is one contiguous 32-byte LDS read (2x ds_load_b128).
__global__ __launch_bounds__(256) void k_xw_wmma(const float* __restrict__ x,
                                                 const float* __restrict__ W,
                                                 float* __restrict__ xw, int N) {
    // Blds[kTile][colTile][lane][i] : per-lane WMMA B fragments, 32B contiguous per lane
    __shared__ __attribute__((aligned(32))) _Float16 Blds[4][4][32][16];  // 16 KB

    int tid = threadIdx.x;

#if __has_builtin(__builtin_amdgcn_global_load_async_to_lds_b128)
    __shared__ __attribute__((aligned(16))) float Wstage[DIMK * HC];      // 25.6 KB raw f32
    // async DMA: 1600 x 16B chunks of W into LDS, tracked by ASYNCcnt
    for (int i = tid; i < (DIMK * HC) / 4; i += 256) {
        __builtin_amdgcn_global_load_async_to_lds_b128(
            (v4i*)(W + 4 * i),
            (v4i*)(&Wstage[4 * i]),
            0, 0);
    }
#if __has_builtin(__builtin_amdgcn_s_wait_asynccnt)
    __builtin_amdgcn_s_wait_asynccnt(0);
#else
    asm volatile("s_wait_asynccnt 0x0" ::: "memory");
#endif
    __syncthreads();
#define WSRC(k, c) Wstage[(k) * HC + (c)]
#else
#define WSRC(k, c) W[(k) * HC + (c)]
#endif

    // swizzle + convert into B-fragment layout:
    //   value(kkt,t,lane,i) = W[K][col], K = kkt*32 + (lane<16?0:8) + (i<8 ? i : 8+i),
    //                         col = t*16 + (lane&15)   (zero-pad col>=50)
    for (int idx = tid; idx < 4 * 4 * 32 * 16; idx += 256) {
        int i    = idx & 15;
        int lane = (idx >> 4) & 31;
        int t    = (idx >> 9) & 3;
        int kkt  = idx >> 11;
        int col  = t * 16 + (lane & 15);
        int K    = kkt * 32 + ((lane < 16) ? 0 : 8) + ((i < 8) ? i : 8 + i);
        float v  = (col < HC) ? WSRC(K, col) : 0.0f;
        Blds[kkt][t][lane][i] = (_Float16)v;
    }
    __syncthreads();

    int wave    = blockIdx.x * 8 + (tid >> 5);
    int lane    = tid & 31;
    int rowBase = wave * 16;
    if (rowBase >= N) return;                 // wave-uniform; EXEC stays all-ones

    int laneLo = lane & 15;
    int row    = rowBase + laneLo;
    if (row >= N) row = N - 1;                // safety clamp (N%16==0 in practice)
    int koffA  = (lane < 16) ? 0 : 8;         // ISA 16-bit A 16x32 layout

    v8f acc[4] = {};

    #pragma unroll
    for (int kkt = 0; kkt < 4; ++kkt) {
        const float* xrow = x + (size_t)row * DIMK + kkt * 32;
        // prefetch the next K-slab of this row (global_prefetch_b8; speculative)
        __builtin_prefetch(xrow + 32, 0, 0);

        // A fragment: elements 0..7 -> K = koffA+0..7 ; 8..15 -> K = koffA+16..23
        const float* p = xrow + koffA;
        v16h a;
        #pragma unroll
        for (int i = 0; i < 8; ++i) a[i]     = (_Float16)p[i];
        #pragma unroll
        for (int i = 0; i < 8; ++i) a[8 + i] = (_Float16)p[16 + i];

        #pragma unroll
        for (int t = 0; t < 4; ++t) {
            v16h b = *(const v16h*)&Blds[kkt][t][lane][0];   // 2x ds_load_b128
            acc[t] = __builtin_amdgcn_wmma_f32_16x16x32_f16(
                false, a, false, b, (short)0, acc[t], false, false);
        }
    }

    // D layout: VGPR r -> row r (lanes 0-15) / row r+8 (lanes 16-31), col = lane%16
    int mAdd = (lane < 16) ? 0 : 8;
    #pragma unroll
    for (int t = 0; t < 4; ++t) {
        int col = t * 16 + laneLo;
        if (col < HC) {
            #pragma unroll
            for (int r = 0; r < 8; ++r) {
                int m = rowBase + r + mAdd;
                if (m < N) xw[(size_t)m * HC + col] = acc[t][r];
            }
        }
    }
}

// ---------------- kernel 2: per-node attention coefficients ----------------
__global__ __launch_bounds__(256) void k_attn_coeff(const float* __restrict__ xw,
                                                    const float* __restrict__ att_src,
                                                    const float* __restrict__ att_dst,
                                                    float* __restrict__ a_src,
                                                    float* __restrict__ a_dst, int N) {
    int n = blockIdx.x * blockDim.x + threadIdx.x;
    if (n >= N) return;
    const float* row = xw + (size_t)n * HC;
    #pragma unroll
    for (int h = 0; h < NH; ++h) {
        float s = 0.0f, d = 0.0f;
        #pragma unroll
        for (int c = 0; c < NC; ++c) {
            float v = row[h * NC + c];
            s += v * att_src[h * NC + c];
            d += v * att_dst[h * NC + c];
        }
        a_src[n * NH + h] = s;
        a_dst[n * NH + h] = d;
    }
}

// ---------------- kernel 3: segment max of edge logits ----------------
__global__ __launch_bounds__(256) void k_edge_max(const int* __restrict__ srcA,
                                                  const int* __restrict__ dstA,
                                                  const float* __restrict__ a_src,
                                                  const float* __restrict__ a_dst,
                                                  float* __restrict__ mmax,
                                                  int E, int Etot) {
    int e = blockIdx.x * blockDim.x + threadIdx.x;
    if (e >= Etot) return;
    int s = (e < E) ? srcA[e] : (e - E);   // self-loops appended
    int d = (e < E) ? dstA[e] : (e - E);
    #pragma unroll
    for (int h = 0; h < NH; ++h) {
        float l = leaky_relu(a_src[s * NH + h] + a_dst[d * NH + h]);
        atomicMaxFloat(&mmax[d * NH + h], l);
    }
}

// ---------------- kernel 4: softmax denominators ----------------
__global__ __launch_bounds__(256) void k_edge_denom(const int* __restrict__ srcA,
                                                    const int* __restrict__ dstA,
                                                    const float* __restrict__ a_src,
                                                    const float* __restrict__ a_dst,
                                                    const float* __restrict__ mmax,
                                                    float* __restrict__ denom,
                                                    int E, int Etot) {
    int e = blockIdx.x * blockDim.x + threadIdx.x;
    if (e >= Etot) return;
    int s = (e < E) ? srcA[e] : (e - E);
    int d = (e < E) ? dstA[e] : (e - E);
    #pragma unroll
    for (int h = 0; h < NH; ++h) {
        float l = leaky_relu(a_src[s * NH + h] + a_dst[d * NH + h]);
        atomicAdd(&denom[d * NH + h], __expf(l - mmax[d * NH + h]));
    }
}

// ---------------- kernel 5: alpha-weighted message scatter ----------------
__global__ __launch_bounds__(256) void k_edge_scatter(const int* __restrict__ srcA,
                                                      const int* __restrict__ dstA,
                                                      const float* __restrict__ a_src,
                                                      const float* __restrict__ a_dst,
                                                      const float* __restrict__ mmax,
                                                      const float* __restrict__ denom,
                                                      const float* __restrict__ xw,
                                                      float* __restrict__ acc,
                                                      int E, int Etot) {
    int e = blockIdx.x * blockDim.x + threadIdx.x;
    if (e >= Etot) return;
    int s = (e < E) ? srcA[e] : (e - E);
    int d = (e < E) ? dstA[e] : (e - E);
    const float* xs = xw + (size_t)s * HC;
    float* ad = acc + (size_t)d * HC;
    #pragma unroll
    for (int h = 0; h < NH; ++h) {
        float l = leaky_relu(a_src[s * NH + h] + a_dst[d * NH + h]);
        float alpha = __expf(l - mmax[d * NH + h]) / denom[d * NH + h];
        #pragma unroll
        for (int c = 0; c < NC; ++c) {
            atomicAdd(&ad[h * NC + c], alpha * xs[h * NC + c]);
        }
    }
}

// ---------------- kernel 6: bias + ELU + graph pooling ----------------
__global__ __launch_bounds__(256) void k_node_finish(const float* __restrict__ acc,
                                                     const float* __restrict__ bias,
                                                     const int* __restrict__ batch,
                                                     float* __restrict__ gsum,
                                                     float* __restrict__ cnt, int N) {
    int n = blockIdx.x * blockDim.x + threadIdx.x;
    if (n >= N) return;
    int g = batch[n];
    const float* an = acc + (size_t)n * HC;
    #pragma unroll
    for (int j = 0; j < HC; ++j) {
        float v = an[j] + bias[j];
        v = v > 0.0f ? v : (__expf(v) - 1.0f);   // ELU
        atomicAdd(&gsum[g * HC + j], v);
    }
    atomicAdd(&cnt[g], 1.0f);
}

// ---------------- kernel 7: mean pool + linear head + sigmoid ----------------
__global__ __launch_bounds__(256) void k_graph_head(const float* __restrict__ gsum,
                                                    const float* __restrict__ cnt,
                                                    const float* __restrict__ lin_w,
                                                    const float* __restrict__ lin_b,
                                                    float* __restrict__ out) {
    int g = blockIdx.x * blockDim.x + threadIdx.x;
    if (g >= NGRAPH) return;
    float inv = 1.0f / fmaxf(cnt[g], 1.0f);
    float dotv = lin_b[0];
    #pragma unroll
    for (int j = 0; j < HC; ++j) {
        float hv = gsum[g * HC + j] * inv;
        out[g * HC + j] = hv;                // h : [G, HC] flattened
        dotv += hv * lin_w[j];
    }
    out[NGRAPH * HC + g] = 1.0f / (1.0f + __expf(-dotv));   // y : [G, 1]
}

// ---------------- host launcher ----------------
extern "C" void kernel_launch(void* const* d_in, const int* in_sizes, int n_in,
                              void* d_out, int out_size, void* d_ws, size_t ws_size,
                              hipStream_t stream) {
    const float* x        = (const float*)d_in[0];
    const float* W        = (const float*)d_in[1];
    const float* att_src  = (const float*)d_in[2];
    const float* att_dst  = (const float*)d_in[3];
    const float* bias     = (const float*)d_in[4];
    const float* lin_w    = (const float*)d_in[5];
    const float* lin_b    = (const float*)d_in[6];
    const int*   edge_idx = (const int*)d_in[7];
    const int*   batch    = (const int*)d_in[8];

    const int N    = in_sizes[8];          // nodes
    const int E    = in_sizes[7] / 2;      // edges (pre self-loop)
    const int Etot = E + N;                // edges + self loops

    const int* srcA = edge_idx;            // edge_index[0]
    const int* dstA = edge_idx + E;        // edge_index[1]

    // carve scratch out of d_ws (all f32)
    float* xw    = (float*)d_ws;                 // N*HC
    float* a_src = xw    + (size_t)N * HC;       // N*NH
    float* a_dst = a_src + (size_t)N * NH;       // N*NH
    float* mmax  = a_dst + (size_t)N * NH;       // N*NH
    float* denom = mmax  + (size_t)N * NH;       // N*NH
    float* acc   = denom + (size_t)N * NH;       // N*HC
    float* gsum  = acc   + (size_t)N * HC;       // G*HC
    float* cnt   = gsum  + (size_t)NGRAPH * HC;  // G

    float* out = (float*)d_out;

    const int TB = 256;
    int initTot = N * HC;                        // covers every init range
    k_init<<<(initTot + TB - 1) / TB, TB, 0, stream>>>(mmax, denom, acc, gsum, cnt, N);

    int waves   = (N + 15) / 16;
    int gemmBlk = (waves + 7) / 8;
    k_xw_wmma<<<gemmBlk, TB, 0, stream>>>(x, W, xw, N);

    k_attn_coeff<<<(N + TB - 1) / TB, TB, 0, stream>>>(xw, att_src, att_dst, a_src, a_dst, N);

    int eBlk = (Etot + TB - 1) / TB;
    k_edge_max    <<<eBlk, TB, 0, stream>>>(srcA, dstA, a_src, a_dst, mmax, E, Etot);
    k_edge_denom  <<<eBlk, TB, 0, stream>>>(srcA, dstA, a_src, a_dst, mmax, denom, E, Etot);
    k_edge_scatter<<<eBlk, TB, 0, stream>>>(srcA, dstA, a_src, a_dst, mmax, denom, xw, acc, E, Etot);

    k_node_finish<<<(N + TB - 1) / TB, TB, 0, stream>>>(acc, bias, batch, gsum, cnt, N);
    k_graph_head<<<1, TB, 0, stream>>>(gsum, cnt, lin_w, lin_b, out);
}
